// MambaBlock_81183471829337
// MI455X (gfx1250) — matrix operand order repaired and checked by reading, not compiled, not run on hardware
//
#include <hip/hip_runtime.h>
#include <hip/hip_bf16.h>

// ---------------------------------------------------------------------------
// Mamba block forward for MI455X (gfx1250, wave32, WMMA).
// D_MODEL=1024 D_STATE=16 D_CONV=4 D_INNER=2048 DT_RANK=64 BATCH=2 SEQ=4096
// GEMMs run on the matrix pipe (v_wmma_f32_16x16x32_f16, f32 accumulate)
// with all operands pre-converted to f16 so the inner loop is loads + WMMA.
// ---------------------------------------------------------------------------

#define D_MODEL 1024
#define D_STATE 16
#define D_CONV  4
#define D_INNER 2048
#define DT_RANK 64
#define BATCH   2
#define SEQ     4096
#define NTOK    (BATCH * SEQ)   // 8192 tokens

typedef __attribute__((ext_vector_type(16))) _Float16 v16h;
typedef __attribute__((ext_vector_type(8)))  _Float16 h8;
typedef __attribute__((ext_vector_type(8)))  float    v8f;
typedef __attribute__((ext_vector_type(4)))  float    f4;

// ---------------------------------------------------------------------------
// WMMA fragment loaders (f16 memory -> f16 fragments, b128 loads).
// A fragment 16x32 (MxK): lanes 0-15 hold M=lane with K={0..7,16..23};
//   lanes 16-31 same rows, K offset +8.  => two 8-half contiguous runs.
// B fragment 32x16 (KxN): lane n%16 holds column n; lanes 0-15: K={0..7,16..23},
//   lanes 16-31: K offset +8 (same as A).
// ---------------------------------------------------------------------------
__device__ __forceinline__ v16h load_a_frag(const _Float16* __restrict__ A, int lda,
                                            int row, int kb, int half) {
    const _Float16* p = A + (size_t)row * lda + kb + half * 8;
    h8 lo = *(const h8*)(p);
    h8 hi = *(const h8*)(p + 16);
    return __builtin_shufflevector(lo, hi, 0,1,2,3,4,5,6,7,8,9,10,11,12,13,14,15);
}

__device__ __forceinline__ v16h load_b_frag(const _Float16* __restrict__ W, int ldw,
                                            int col, int kb, int half) {
    const _Float16* p = W + (size_t)col * ldw + kb + half * 8;
    h8 lo = *(const h8*)(p);
    h8 hi = *(const h8*)(p + 16);
    return __builtin_shufflevector(lo, hi, 0,1,2,3,4,5,6,7,8,9,10,11,12,13,14,15);
}

__device__ __forceinline__ float softplus_f(float v) {
    return v > 20.f ? v : __logf(1.f + __expf(v));
}
__device__ __forceinline__ float silu_f(float v) {
    return v / (1.f + __expf(-v));
}

// ---------------------------------------------------------------------------
// WMMA GEMM: C[m,n] = act( sum_k A[m,k] * W[n,k] + bias[n] ) + res[m,n]
// A: M x K row-major f16 (stride lda), W: N x K row-major f16 (stride ldw).
// Optional secondary f16 output C16 (same layout as C).
// Block = 128 threads = 4 waves; wave tile 64x64 (4x4 WMMA subtiles);
// block tile 128x128. K multiple of 32; M multiple of 128.
// act: 0 = identity, 1 = softplus.
// ---------------------------------------------------------------------------
__global__ void __launch_bounds__(128)
gemm_wmma_f16(const _Float16* __restrict__ A, int lda,
              const _Float16* __restrict__ W, int ldw,
              float* __restrict__ C, _Float16* __restrict__ C16, int ldc,
              int M, int N, int K,
              const float* __restrict__ bias,
              const float* __restrict__ residual, int ldr,
              int act) {
    const int lane = threadIdx.x & 31;
    const int wave = threadIdx.x >> 5;
    const int half = lane >> 4;
    const int l16  = lane & 15;

    const int mBase = blockIdx.x * 128 + (wave >> 1) * 64;
    const int nBase = blockIdx.y * 128 + (wave & 1) * 64;

    v8f acc[4][4];
#pragma unroll
    for (int i = 0; i < 4; ++i)
#pragma unroll
        for (int j = 0; j < 4; ++j)
#pragma unroll
            for (int r = 0; r < 8; ++r) acc[i][j][r] = 0.f;

    for (int kb = 0; kb < K; kb += 32) {
        v16h a[4], b[4];
#pragma unroll
        for (int i = 0; i < 4; ++i) {
            a[i] = load_a_frag(A, lda, mBase + i * 16 + l16, kb, half);
        }
#pragma unroll
        for (int j = 0; j < 4; ++j) {
            int col = nBase + j * 16 + l16;
            if (col > N - 1) col = N - 1;   // clamp (guarded at store)
            b[j] = load_b_frag(W, ldw, col, kb, half);
        }
        if (kb + 32 < K) {
            __builtin_prefetch(A + (size_t)(mBase + l16) * lda + kb + 32, 0, 1);
        }
#pragma unroll
        for (int i = 0; i < 4; ++i)
#pragma unroll
            for (int j = 0; j < 4; ++j)
                acc[i][j] = __builtin_amdgcn_wmma_f32_16x16x32_f16(
                    false, a[i], false, b[j], (short)0, acc[i][j], false, false);
    }

    // Epilogue: C layout — lane l16 holds column n; VGPR r holds row r + half*8.
#pragma unroll
    for (int j = 0; j < 4; ++j) {
        const int n = nBase + j * 16 + l16;
        if (n >= N) continue;
        const float bv = bias ? bias[n] : 0.f;
#pragma unroll
        for (int i = 0; i < 4; ++i) {
#pragma unroll
            for (int r = 0; r < 8; ++r) {
                const int m = mBase + i * 16 + r + half * 8;
                float v = acc[i][j][r] + bv;
                if (act == 1) v = softplus_f(v);
                if (residual) v += residual[(size_t)m * ldr + n];
                if (C)   C[(size_t)m * ldc + n] = v;
                if (C16) C16[(size_t)m * ldc + n] = (_Float16)v;
            }
        }
    }
}

// ---------------------------------------------------------------------------
// f32 -> f16 conversion (weights), vectorized 4-wide. n multiple of 4.
// ---------------------------------------------------------------------------
typedef __attribute__((ext_vector_type(4))) _Float16 h4;
__global__ void __launch_bounds__(256)
cvt_f32_f16_kernel(const float* __restrict__ src, _Float16* __restrict__ dst, int n4) {
    const int i = blockIdx.x * 256 + threadIdx.x;
    if (i >= n4) return;
    f4 v = ((const f4*)src)[i];
    h4 h;
    h[0] = (_Float16)v[0]; h[1] = (_Float16)v[1];
    h[2] = (_Float16)v[2]; h[3] = (_Float16)v[3];
    ((h4*)dst)[i] = h;
}

// ---------------------------------------------------------------------------
// RMSNorm: one block per token, 256 threads. Writes f16 (GEMM A operand).
// ---------------------------------------------------------------------------
__global__ void __launch_bounds__(256)
rmsnorm_kernel(const float* __restrict__ x, const float* __restrict__ w,
               _Float16* __restrict__ xn16) {
    const int t = blockIdx.x;
    const float* px = x + (size_t)t * D_MODEL;
    float s = 0.f;
    for (int i = threadIdx.x; i < D_MODEL; i += 256) {
        float v = px[i];
        s += v * v;
    }
#pragma unroll
    for (int off = 16; off > 0; off >>= 1) s += __shfl_xor(s, off, 32);
    __shared__ float sm[8];
    if ((threadIdx.x & 31) == 0) sm[threadIdx.x >> 5] = s;
    __syncthreads();
    float tot = 0.f;
#pragma unroll
    for (int k = 0; k < 8; ++k) tot += sm[k];
    const float scale = rsqrtf(tot * (1.f / D_MODEL) + 1e-6f);
    for (int i = threadIdx.x; i < D_MODEL; i += 256) {
        xn16[(size_t)t * D_MODEL + i] = (_Float16)(px[i] * scale * w[i]);
    }
}

// ---------------------------------------------------------------------------
// Causal depthwise conv1d (width 4) + SiLU.
// x_b is the first D_INNER columns of xz (row stride 2*D_INNER).
// Writes f32 (scan input) and f16 (x_proj GEMM A operand).
// ---------------------------------------------------------------------------
__global__ void __launch_bounds__(256)
conv_silu_kernel(const float* __restrict__ xz, const float* __restrict__ cw,
                 const float* __restrict__ cb,
                 float* __restrict__ xc, _Float16* __restrict__ xc16) {
    const size_t idx = (size_t)blockIdx.x * 256 + threadIdx.x;
    if (idx >= (size_t)NTOK * D_INNER) return;
    const int d = (int)(idx % D_INNER);
    const int t = (int)(idx / D_INNER);
    const int l = t % SEQ;
    const int b = t / SEQ;
    float s = cb[d];
#pragma unroll
    for (int j = 0; j < D_CONV; ++j) {
        const int ll = l - (D_CONV - 1) + j;
        if (ll >= 0) {
            s += cw[d * D_CONV + j] *
                 xz[((size_t)b * SEQ + ll) * (2 * D_INNER) + d];
        }
    }
    const float v = silu_f(s);
    xc[idx] = v;
    xc16[idx] = (_Float16)v;
}

// ---------------------------------------------------------------------------
// Selective scan: one lane per (b, d, s). 16 state lanes per channel,
// cross-lane sum via shfl_xor(width=16). Fuses y = (h.C + D*x) * silu(z).
// Writes f16 (out_proj GEMM A operand).
// ---------------------------------------------------------------------------
__global__ void __launch_bounds__(256)
scan_kernel(const float* __restrict__ xc, const float* __restrict__ dt,
            const float* __restrict__ xdbc, const float* __restrict__ A_log,
            const float* __restrict__ Dvec, const float* __restrict__ xz,
            _Float16* __restrict__ y16) {
    const int gid = blockIdx.x * 256 + threadIdx.x;
    const int s   = gid & (D_STATE - 1);
    const int ch  = gid >> 4;              // (b, d) channel
    const int d   = ch & (D_INNER - 1);
    const int b   = ch >> 11;              // / D_INNER
    const float As = -__expf(A_log[d * D_STATE + s]);
    const float Dd = Dvec[d];

    float h = 0.f;
    const size_t tbase = (size_t)b * SEQ;
    for (int l = 0; l < SEQ; ++l) {
        const size_t t = tbase + l;
        if (l + 8 < SEQ) {
            __builtin_prefetch(xc + (t + 8) * D_INNER + d, 0, 1);
            __builtin_prefetch(dt + (t + 8) * D_INNER + d, 0, 1);
            __builtin_prefetch(xdbc + (t + 8) * 96 + DT_RANK + s, 0, 1);
        }
        const float xt  = xc[t * D_INNER + d];
        const float dtt = dt[t * D_INNER + d];
        const float Bs  = xdbc[t * 96 + DT_RANK + s];
        const float Cs  = xdbc[t * 96 + DT_RANK + D_STATE + s];
        const float dA  = __expf(dtt * As);
        h = dA * h + (dtt * xt) * Bs;
        float v = h * Cs;
#pragma unroll
        for (int off = 8; off > 0; off >>= 1) v += __shfl_xor(v, off, 16);
        if (s == 0) {
            const float zt = xz[t * (2 * D_INNER) + D_INNER + d];
            y16[t * D_INNER + d] = (_Float16)((v + Dd * xt) * silu_f(zt));
        }
    }
}

// ---------------------------------------------------------------------------
// Launch
// ---------------------------------------------------------------------------
extern "C" void kernel_launch(void* const* d_in, const int* in_sizes, int n_in,
                              void* d_out, int out_size, void* d_ws, size_t ws_size,
                              hipStream_t stream) {
    const float* x          = (const float*)d_in[0];
    const float* norm_w     = (const float*)d_in[1];
    const float* in_proj_w  = (const float*)d_in[2];   // (4096, 1024)
    const float* conv_w     = (const float*)d_in[3];   // (2048, 1, 4)
    const float* conv_b     = (const float*)d_in[4];
    const float* x_proj_w   = (const float*)d_in[5];   // (96, 2048)
    const float* dt_proj_w  = (const float*)d_in[6];   // (2048, 64)
    const float* dt_proj_b  = (const float*)d_in[7];
    const float* A_log      = (const float*)d_in[8];   // (2048, 16)
    const float* Dvec       = (const float*)d_in[9];
    const float* out_proj_w = (const float*)d_in[10];  // (1024, 2048)
    float* out = (float*)d_out;

    // ---- workspace layout -------------------------------------------------
    float* ws = (float*)d_ws;
    size_t off = 0;                                    // in floats
    float* xz   = ws + off; off += (size_t)NTOK * 2 * D_INNER;    // 8192x4096 f32
    float* xc   = ws + off; off += (size_t)NTOK * D_INNER;        // 8192x2048 f32
    float* xdbc = ws + off; off += (size_t)NTOK * 96;             // 8192x96   f32
    float* dt   = ws + off; off += (size_t)NTOK * D_INNER;        // 8192x2048 f32

    _Float16* hbase = (_Float16*)(ws + off);
    size_t hoff = 0;                                   // in halves
    _Float16* xn16   = hbase + hoff; hoff += (size_t)NTOK * D_MODEL;
    _Float16* xc16   = hbase + hoff; hoff += (size_t)NTOK * D_INNER;
    _Float16* xdbc16 = hbase + hoff; hoff += (size_t)NTOK * 96;
    _Float16* y16    = hbase + hoff; hoff += (size_t)NTOK * D_INNER;
    _Float16* inw16  = hbase + hoff; hoff += (size_t)(2 * D_INNER) * D_MODEL;
    _Float16* xpw16  = hbase + hoff; hoff += (size_t)96 * D_INNER;
    _Float16* dtw16  = hbase + hoff; hoff += (size_t)D_INNER * DT_RANK;
    _Float16* outw16 = hbase + hoff; hoff += (size_t)D_MODEL * D_INNER;

    // ---- 0) weight f32 -> f16 conversions (tiny, bandwidth-trivial) -------
    {
        int n4;
        n4 = (2 * D_INNER * D_MODEL) / 4;
        cvt_f32_f16_kernel<<<(n4 + 255) / 256, 256, 0, stream>>>(in_proj_w, inw16, n4);
        n4 = (96 * D_INNER) / 4;
        cvt_f32_f16_kernel<<<(n4 + 255) / 256, 256, 0, stream>>>(x_proj_w, xpw16, n4);
        n4 = (D_INNER * DT_RANK) / 4;
        cvt_f32_f16_kernel<<<(n4 + 255) / 256, 256, 0, stream>>>(dt_proj_w, dtw16, n4);
        n4 = (D_MODEL * D_INNER) / 4;
        cvt_f32_f16_kernel<<<(n4 + 255) / 256, 256, 0, stream>>>(out_proj_w, outw16, n4);
    }

    // ---- 1) RMSNorm -> xn16 ----------------------------------------------
    rmsnorm_kernel<<<NTOK, 256, 0, stream>>>(x, norm_w, xn16);

    // ---- 2) in_proj: (8192x1024) x (4096x1024)^T -> xz f32 ---------------
    gemm_wmma_f16<<<dim3(NTOK / 128, (2 * D_INNER) / 128), 128, 0, stream>>>(
        xn16, D_MODEL, inw16, D_MODEL, xz, nullptr, 2 * D_INNER,
        NTOK, 2 * D_INNER, D_MODEL, nullptr, nullptr, 0, 0);

    // ---- 3) causal depthwise conv + SiLU -> xc f32, xc16 ------------------
    {
        const size_t n = (size_t)NTOK * D_INNER;
        conv_silu_kernel<<<(unsigned)((n + 255) / 256), 256, 0, stream>>>(
            xz, conv_w, conv_b, xc, xc16);
    }

    // ---- 4) x_proj: (8192x2048) x (96x2048)^T -> xdbc f32 + f16 -----------
    gemm_wmma_f16<<<dim3(NTOK / 128, 1), 128, 0, stream>>>(
        xc16, D_INNER, xpw16, D_INNER, xdbc, xdbc16, 96,
        NTOK, 96, D_INNER, nullptr, nullptr, 0, 0);

    // ---- 5) dt_proj + bias + softplus: A = xdbc16[:, :64] (lda=96) --------
    gemm_wmma_f16<<<dim3(NTOK / 128, D_INNER / 128), 128, 0, stream>>>(
        xdbc16, 96, dtw16, DT_RANK, dt, nullptr, D_INNER,
        NTOK, D_INNER, DT_RANK, dt_proj_b, nullptr, 0, 1);

    // ---- 6) selective scan (+ silu(z) gate) -> y16 ------------------------
    scan_kernel<<<(BATCH * D_INNER * D_STATE) / 256, 256, 0, stream>>>(
        xc, dt, xdbc, A_log, Dvec, xz, y16);

    // ---- 7) out_proj + residual: (8192x2048) x (1024x2048)^T + x -> out ---
    gemm_wmma_f16<<<dim3(NTOK / 128, D_MODEL / 128), 128, 0, stream>>>(
        y16, D_INNER, outw16, D_INNER, out, nullptr, D_MODEL,
        NTOK, D_MODEL, D_INNER, nullptr, x, D_MODEL, 0);
}
